// RANet_87960930222319
// MI455X (gfx1250) — matrix operand order repaired and checked by reading, not compiled, not run on hardware
//
#include <hip/hip_runtime.h>
#include <hip/hip_bf16.h>

#define NN    8192      // nodes
#define FIN   128       // input features
#define DM    512       // model dim
#define HH    1024      // hidden dim
#define EE    131072    // edges
#define NPD   128      // nodes per doc
#define PAIRS 32

typedef __attribute__((ext_vector_type(16))) __bf16 v16bf;
typedef __attribute__((ext_vector_type(8)))  float  v8f;
typedef unsigned short ushort_t;

__device__ __forceinline__ unsigned short f2bf(float x) {
    unsigned int u = __float_as_uint(x);
    u += 0x7fffu + ((u >> 16) & 1u);           // round-to-nearest-even
    return (unsigned short)(u >> 16);
}

// ---------------------------------------------------------------------------
// WMMA GEMM: C[M,N] = act(A[M,K] @ B[K,N] + bias)
//   A  : M x K bf16 row-major
//   Bt : N x K bf16 row-major (pre-transposed weight)
//   out: Cb != null -> bf16 store; else f32 store to Cf
// tile: 128x128 per 256-thread block; 8 waves (4x2), each wave 32x64
//       (2 A-frags x 4 B-frags = 8 wmma per K-step), double-buffered LDS
// requires: M%128==0, N%128==0, K%32==0
// act: 0 = none, 1 = relu, 2 = tanh
// ---------------------------------------------------------------------------
#define LDA 40
#define LDB 40
#define ABUF (128 * LDA)
#define BBUF (128 * LDB)

__global__ __launch_bounds__(256) void wmma_gemm_kernel(
    const ushort_t* __restrict__ A, const ushort_t* __restrict__ Bt,
    const float* __restrict__ bias,
    float* __restrict__ Cf, ushort_t* __restrict__ Cb,
    int M, int N, int K, int act)
{
    __shared__ __align__(16) ushort_t As[2 * ABUF];
    __shared__ __align__(16) ushort_t Bs[2 * BBUF];

    const int tid  = threadIdx.x;
    const int lane = tid & 31;
    const int wave = tid >> 5;
    const int wm   = wave & 3;        // 0..3  (M direction, 32 rows each)
    const int wn   = wave >> 2;       // 0..1  (N direction, 64 cols each)
    const int L    = lane & 15;
    const int h    = lane >> 4;
    const long long blockM = (long long)blockIdx.y * 128;
    const long long blockN = (long long)blockIdx.x * 128;

    v8f acc[2][4];
    v8f vzero = {0.f,0.f,0.f,0.f,0.f,0.f,0.f,0.f};
#pragma unroll
    for (int mi = 0; mi < 2; ++mi)
#pragma unroll
        for (int ni = 0; ni < 4; ++ni) acc[mi][ni] = vzero;

    // staging: thread loads uint4 (8 bf16, k-contiguous); rows srow, srow+64
    const int srow = tid >> 2;            // 0..63
    const int scol = (tid & 3) * 8;       // 0,8,16,24
    const ushort_t* ap0 = A  + (blockM + srow) * (long long)K + scol;
    const ushort_t* ap1 = ap0 + 64 * (long long)K;
    const ushort_t* bp0 = Bt + (blockN + srow) * (long long)K + scol;
    const ushort_t* bp1 = bp0 + 64 * (long long)K;
    ushort_t* as0 = &As[srow * LDA + scol];
    ushort_t* as1 = as0 + 64 * LDA;
    ushort_t* bs0 = &Bs[srow * LDB + scol];
    ushort_t* bs1 = bs0 + 64 * LDB;

    // prologue: stage tile 0 into buffer 0
    {
        uint4 ra0 = *(const uint4*)ap0, ra1 = *(const uint4*)ap1;
        uint4 rb0 = *(const uint4*)bp0, rb1 = *(const uint4*)bp1;
        ap0 += 32; ap1 += 32; bp0 += 32; bp1 += 32;
        *(uint4*)as0 = ra0; *(uint4*)as1 = ra1;
        *(uint4*)bs0 = rb0; *(uint4*)bs1 = rb1;
    }
    __syncthreads();

    const int nk = K >> 5;
    for (int i = 0; i < nk; ++i) {
        const int cur  = i & 1;
        const bool more = (i + 1 < nk);
        uint4 ra0, ra1, rb0, rb1;
        if (more) {   // fetch next tile while current computes
            ra0 = *(const uint4*)ap0; ra1 = *(const uint4*)ap1;
            rb0 = *(const uint4*)bp0; rb1 = *(const uint4*)bp1;
            __builtin_prefetch(ap0 + 32, 0, 1);   // global_prefetch_b8 (tile i+2)
            __builtin_prefetch(ap1 + 32, 0, 1);
            __builtin_prefetch(bp0 + 32, 0, 1);
            __builtin_prefetch(bp1 + 32, 0, 1);
            ap0 += 32; ap1 += 32; bp0 += 32; bp1 += 32;
        }

        // fragment loads (ISA VGPR layouts)
        // A 16x32: lane(L,h): q0 = K 8h..8h+7, q1 = K 16+8h..+7
        // B 32x16 (as [n][k]): lane(L,h): q0 = K 8h..8h+7, q1 = K 16+8h..+7
        const ushort_t* Ab = &As[cur * ABUF];
        const ushort_t* Bb = &Bs[cur * BBUF];
        union U { v16bf v; uint4 q[2]; };
        U a[2], b[4];
#pragma unroll
        for (int mi = 0; mi < 2; ++mi) {
            const ushort_t* p = &Ab[(wm * 32 + mi * 16 + L) * LDA + 8 * h];
            a[mi].q[0] = *(const uint4*)p;
            a[mi].q[1] = *(const uint4*)(p + 16);
        }
#pragma unroll
        for (int ni = 0; ni < 4; ++ni) {
            const ushort_t* p = &Bb[(wn * 64 + ni * 16 + L) * LDB + 8 * h];
            b[ni].q[0] = *(const uint4*)p;
            b[ni].q[1] = *(const uint4*)(p + 16);
        }
#pragma unroll
        for (int mi = 0; mi < 2; ++mi)
#pragma unroll
            for (int ni = 0; ni < 4; ++ni)
                acc[mi][ni] = __builtin_amdgcn_wmma_f32_16x16x32_bf16(
                    false, a[mi].v, false, b[ni].v, (short)0, acc[mi][ni], false, false);

        if (more) {   // stage next tile into the other buffer
            const int nxt = cur ^ 1;
            *(uint4*)(as0 + nxt * ABUF) = ra0; *(uint4*)(as1 + nxt * ABUF) = ra1;
            *(uint4*)(bs0 + nxt * BBUF) = rb0; *(uint4*)(bs1 + nxt * BBUF) = rb1;
        }
        __syncthreads();
    }

    // ---- epilogue: bias + activation ----
#pragma unroll
    for (int mi = 0; mi < 2; ++mi) {
#pragma unroll
        for (int ni = 0; ni < 4; ++ni) {
            long long n = blockN + wn * 64 + ni * 16 + L;
            float bv = bias ? bias[n] : 0.f;
#pragma unroll
            for (int r = 0; r < 8; ++r) {
                long long m = blockM + wm * 32 + mi * 16 + r + 8 * h;
                float v = acc[mi][ni][r] + bv;
                if (act == 1) v = fmaxf(v, 0.f);
                else if (act == 2) v = tanhf(v);
                if (Cb) Cb[m * (long long)N + n] = f2bf(v);
                else    Cf[m * (long long)N + n] = v;
            }
        }
    }
}

// ---------------------------------------------------------------------------
// conversion kernels (run outside the GEMM hot path)
// ---------------------------------------------------------------------------
__global__ void conv_bf_kernel(const float* __restrict__ src, ushort_t* __restrict__ dst, int n) {
    int i = blockIdx.x * blockDim.x + threadIdx.x;
    if (i < n) dst[i] = f2bf(src[i]);
}

// Bt[n*K + k] = bf16(B[k*N + n])   (weights, once per launch)
__global__ void convT_bf_kernel(const float* __restrict__ B, ushort_t* __restrict__ Bt, int K, int N) {
    int g = blockIdx.x * blockDim.x + threadIdx.x;
    if (g >= K * N) return;
    int k = g / N, n = g - k * N;
    Bt[(long long)n * K + k] = f2bf(B[g]);
}

__global__ void tanh_bf_kernel(const float* __restrict__ src, ushort_t* __restrict__ dst, int n) {
    int i = blockIdx.x * blockDim.x + threadIdx.x;
    if (i < n) dst[i] = f2bf(tanhf(src[i]));
}

// ---------------------------------------------------------------------------
// graph kernels
// ---------------------------------------------------------------------------
__global__ void zero_kernel(float* p, int n) {
    int i = blockIdx.x * blockDim.x + threadIdx.x;
    if (i < n) p[i] = 0.f;
}

__global__ void deg_kernel(const int* __restrict__ col, float* __restrict__ deg, int e) {
    int i = blockIdx.x * blockDim.x + threadIdx.x;
    if (i < e) atomicAdd(&deg[col[i]], 1.f);
}

__global__ void dinv_kernel(float* d, int n) {
    int i = blockIdx.x * blockDim.x + threadIdx.x;
    if (i < n) d[i] = rsqrtf(d[i] + 1.f);
}

// G[i,f] = dinv[i]^2 * XW[i,f] + b[f]
__global__ void gcn_init_kernel(const float* __restrict__ XW, const float* __restrict__ dinv,
                                const float* __restrict__ b, float* __restrict__ G) {
    int g = blockIdx.x * blockDim.x + threadIdx.x;   // < NN*DM
    int i = g >> 9, f = g & 511;
    float s = dinv[i];
    G[g] = s * s * XW[g] + b[f];
}

// scatter: G[col,f] += dinv[row]*dinv[col] * XW[row,f]   (4 floats per thread)
__global__ void gcn_edge_kernel(const float* __restrict__ XW, const int* __restrict__ row,
                                const int* __restrict__ col, const float* __restrict__ dinv,
                                float* __restrict__ G) {
    int g = blockIdx.x * blockDim.x + threadIdx.x;   // < EE * (DM/4)
    int e = g >> 7;
    int c = (g & 127) * 4;
    int r = row[e], d = col[e];
    float ne = dinv[r] * dinv[d];
    float4 v = *(const float4*)&XW[(long long)r * DM + c];
    float* dst = &G[(long long)d * DM + c];
    atomicAdd(dst + 0, ne * v.x);
    atomicAdd(dst + 1, ne * v.y);
    atomicAdd(dst + 2, ne * v.z);
    atomicAdd(dst + 3, ne * v.w);
}

// one wave per query row; 128 allowed keys (other doc of the pair); bf16 out
__global__ __launch_bounds__(256) void attn_kernel(
    const float* __restrict__ Q, const float* __restrict__ K,
    const float* __restrict__ V, ushort_t* __restrict__ O)
{
    int q    = blockIdx.x * 8 + (threadIdx.x >> 5);   // query node, < NN
    int lane = threadIdx.x & 31;
    int p = q >> 8;                  // pair
    int s = q & 255;                 // position within pair
    int kb = (p << 8) + ((s < NPD) ? NPD : 0);        // other doc's 128 rows
    const float scale = 0.04419417382415922f;         // 1/sqrt(512)

    float qv[16];
#pragma unroll
    for (int j = 0; j < 16; ++j) qv[j] = Q[(long long)q * DM + j * 32 + lane];

    float sc[4];
    for (int t = 0; t < NPD; ++t) {
        const float* kr = K + (long long)(kb + t) * DM;
        float d = 0.f;
#pragma unroll
        for (int j = 0; j < 16; ++j) d += qv[j] * kr[j * 32 + lane];
#pragma unroll
        for (int off = 16; off > 0; off >>= 1) d += __shfl_xor(d, off, 32);
        if ((t & 31) == lane) sc[t >> 5] = d * scale;
    }
    float mx = fmaxf(fmaxf(sc[0], sc[1]), fmaxf(sc[2], sc[3]));
#pragma unroll
    for (int off = 16; off > 0; off >>= 1) mx = fmaxf(mx, __shfl_xor(mx, off, 32));
    float sum = 0.f;
#pragma unroll
    for (int i = 0; i < 4; ++i) { sc[i] = __expf(sc[i] - mx); sum += sc[i]; }
#pragma unroll
    for (int off = 16; off > 0; off >>= 1) sum += __shfl_xor(sum, off, 32);
    float inv = 1.f / sum;

    float accv[16];
#pragma unroll
    for (int j = 0; j < 16; ++j) accv[j] = 0.f;
    for (int t = 0; t < NPD; ++t) {
        float pt = __shfl(sc[t >> 5], t & 31, 32) * inv;
        const float* vr = V + (long long)(kb + t) * DM;
#pragma unroll
        for (int j = 0; j < 16; ++j) accv[j] += pt * vr[j * 32 + lane];
    }
#pragma unroll
    for (int j = 0; j < 16; ++j) O[(long long)q * DM + j * 32 + lane] = f2bf(accv[j]);
}

// fc2(tanh) -> fc3 -> mean-pool contribution per node (one wave per node)
__global__ __launch_bounds__(256) void head_kernel(
    const float* __restrict__ Hf, const float* __restrict__ w2, const float* __restrict__ b2,
    const float* __restrict__ w3, const float* __restrict__ b3, float* __restrict__ util)
{
    int node = blockIdx.x * 8 + (threadIdx.x >> 5);
    int lane = threadIdx.x & 31;
    const float* hp = Hf + (long long)node * DM;
    float a = 0.f;
    for (int f = 0; f < DM; ++f) a += hp[f] * w2[f * 32 + lane];
    a = tanhf(a + b2[lane]);
    float part = a * w3[lane];
#pragma unroll
    for (int off = 16; off > 0; off >>= 1) part += __shfl_xor(part, off, 32);
    if (lane == 0)
        atomicAdd(&util[node >> 7], (part + b3[0]) * (1.f / (float)NPD));
}

__global__ void final_kernel(const float* __restrict__ util, float* __restrict__ out) {
    int p = threadIdx.x;
    if (p < PAIRS) {
        float d = util[2 * p + 1] - util[2 * p];
        out[p] = 1.f / (1.f + __expf(-d));
    }
}

// ---------------------------------------------------------------------------
static inline void gemm(hipStream_t s, const ushort_t* A, const ushort_t* Bt,
                        const float* bias, float* Cf, ushort_t* Cb,
                        int M, int N, int K, int act) {
    dim3 grid(N / 128, M / 128);
    wmma_gemm_kernel<<<grid, 256, 0, s>>>(A, Bt, bias, Cf, Cb, M, N, K, act);
}

extern "C" void kernel_launch(void* const* d_in, const int* in_sizes, int n_in,
                              void* d_out, int out_size, void* d_ws, size_t ws_size,
                              hipStream_t stream) {
    (void)in_sizes; (void)n_in; (void)out_size; (void)ws_size;
    const float* x      = (const float*)d_in[0];
    const float* w_in   = (const float*)d_in[1];
    const float* b_in   = (const float*)d_in[2];
    const float* w_mid  = (const float*)d_in[3];
    const float* b_mid  = (const float*)d_in[4];
    const float* w_out  = (const float*)d_in[5];
    const float* b_out  = (const float*)d_in[6];
    const float* fc1_w  = (const float*)d_in[7];
    const float* fc1_b  = (const float*)d_in[8];
    const float* fc2_w  = (const float*)d_in[9];
    const float* fc2_b  = (const float*)d_in[10];
    const float* fc3_w  = (const float*)d_in[11];
    const float* fc3_b  = (const float*)d_in[12];
    const float* qkv1_w = (const float*)d_in[13];
    const float* qkv1_b = (const float*)d_in[14];
    const float* qkv2_w = (const float*)d_in[15];
    const float* qkv2_b = (const float*)d_in[16];
    const float* qkv3_w = (const float*)d_in[17];
    const float* qkv3_b = (const float*)d_in[18];
    const int*   edge   = (const int*)d_in[19];
    const int*   rowp   = edge;
    const int*   colp   = edge + EE;
    float* out = (float*)d_out;

    const size_t ND = (size_t)NN * DM;   // 4,194,304
    const size_t NH = (size_t)NN * HH;   // 8,388,608
    char*  base = (char*)d_ws;
    size_t off = 0;
    auto allocF = [&](size_t n) { float* p = (float*)(base + off); off += n * 4; return p; };
    auto allocU = [&](size_t n) { ushort_t* p = (ushort_t*)(base + off);
                                  off += ((n * 2 + 255) & ~(size_t)255); return p; };
    // f32 buffers
    float* dinv = allocF(NN);
    float* util = allocF(64);
    float* XW   = allocF(ND);
    float* G    = allocF(ND);
    float* Qb   = allocF(ND);
    float* Kb   = allocF(ND);
    float* Vb   = allocF(ND);
    // bf16 activation buffers
    ushort_t* xbf  = allocU((size_t)NN * FIN);
    ushort_t* Gbf  = allocU(ND);
    ushort_t* Hbf  = allocU(ND);
    ushort_t* T1bf = allocU(NH);
    ushort_t* T2bf = allocU(NH);
    // bf16 transposed weights
    ushort_t* w_inT  = allocU((size_t)FIN * DM);
    ushort_t* w_midT = allocU((size_t)DM * DM);
    ushort_t* w_outT = allocU((size_t)DM * DM);
    ushort_t* fc1T   = allocU((size_t)DM * DM);
    ushort_t* qkv1T  = allocU((size_t)3 * DM * HH);
    ushort_t* qkv2T  = allocU((size_t)3 * HH * HH);
    ushort_t* qkv3T  = allocU((size_t)3 * HH * DM);

    // ---- weight conversion + transpose (once per launch) ----
    auto convT = [&](const float* B, ushort_t* Bt, int K, int N) {
        convT_bf_kernel<<<(K * N + 255) / 256, 256, 0, stream>>>(B, Bt, K, N);
    };
    convT(w_in,  w_inT,  FIN, DM);
    convT(w_mid, w_midT, DM, DM);
    convT(w_out, w_outT, DM, DM);
    convT(fc1_w, fc1T,   DM, DM);
    for (int t = 0; t < 3; ++t) {
        convT(qkv1_w + (size_t)t * DM * HH, qkv1T + (size_t)t * HH * DM, DM, HH);
        convT(qkv2_w + (size_t)t * HH * HH, qkv2T + (size_t)t * HH * HH, HH, HH);
        convT(qkv3_w + (size_t)t * HH * DM, qkv3T + (size_t)t * DM * HH, HH, DM);
    }
    conv_bf_kernel<<<(NN * FIN) / 256, 256, 0, stream>>>(x, xbf, NN * FIN);

    // ---- degrees -> dinv ----
    zero_kernel<<<NN / 256, 256, 0, stream>>>(dinv, NN);
    deg_kernel<<<EE / 256, 256, 0, stream>>>(colp, dinv, EE);
    dinv_kernel<<<NN / 256, 256, 0, stream>>>(dinv, NN);

    const ushort_t* hin = xbf;
    int din = FIN;
    for (int layer = 0; layer < 4; ++layer) {
        const ushort_t* Wt = (layer == 0) ? w_inT : (layer == 3) ? w_outT : w_midT;
        const float*    b  = (layer == 0) ? b_in  : (layer == 3) ? b_out  : b_mid;
        // GCN: XW = h @ W ; G = self + scatter + bias ; tanh -> bf16
        gemm(stream, hin, Wt, nullptr, XW, nullptr, NN, DM, din, 0);
        gcn_init_kernel<<<(int)(ND / 256), 256, 0, stream>>>(XW, dinv, b, G);
        gcn_edge_kernel<<<EE * (DM / 4) / 256, 256, 0, stream>>>(XW, rowp, colp, dinv, G);
        tanh_bf_kernel<<<(int)(ND / 256), 256, 0, stream>>>(G, Gbf, (int)ND);
        // QKV three-layer MLPs (relu) via WMMA; T1/T2 stay bf16
        float* qkv[3] = { Qb, Kb, Vb };
        for (int t = 0; t < 3; ++t) {
            gemm(stream, Gbf,  qkv1T + (size_t)t * HH * DM, qkv1_b + (size_t)t * HH, nullptr, T1bf, NN, HH, DM, 1);
            gemm(stream, T1bf, qkv2T + (size_t)t * HH * HH, qkv2_b + (size_t)t * HH, nullptr, T2bf, NN, HH, HH, 1);
            gemm(stream, T2bf, qkv3T + (size_t)t * DM * HH, qkv3_b + (size_t)t * DM, qkv[t], nullptr, NN, DM, HH, 1);
        }
        attn_kernel<<<NN / 8, 256, 0, stream>>>(Qb, Kb, Vb, Hbf);
        hin = Hbf;
        din = DM;
    }

    // head: fc1 (tanh) via WMMA -> f32, then fc2/fc3 + mean-pool + sigmoid
    gemm(stream, Hbf, fc1T, fc1_b, G, nullptr, NN, DM, DM, 2);
    zero_kernel<<<1, 64, 0, stream>>>(util, 64);
    head_kernel<<<NN / 8, 256, 0, stream>>>(G, fc2_w, fc2_b, fc3_w, fc3_b, util);
    final_kernel<<<1, 32, 0, stream>>>(util, out);
}
